// Block_16518444220532
// MI455X (gfx1250) — matrix operand order repaired and checked by reading, not compiled, not run on hardware
//
#include <hip/hip_runtime.h>

#define C_IN   128
#define C_OUT  128
#define KOFF   27
#define EPS_BN 1e-4f
#define LEAK   0.333f

typedef __attribute__((ext_vector_type(16))) __bf16 bf16x16;
typedef __attribute__((ext_vector_type(8)))  float  floatx8;

// ws layout (bytes):
//   [0,    512)    float sum[128]
//   [512,  1024)   float sumsq[128]
//   [1024, 1536)   zero row (512 B zeros; used as f32[128] or bf16 row)
//   [2048, 2048 + 27*128*128*2)  __bf16 Wt[k][cout][cin]   (886,784 end)
//   [1 MB, 1 MB + N*128*2)       __bf16 feat_bf16 (optional, if ws allows)
#define WS_SUM    0
#define WS_ZROW   1024
#define WS_WT     2048
#define WS_FEATB  (1u << 20)

// LDS row stride: 136 bf16 = 272 B = 68 dwords; 68 % 64 = 4 banks skew/row
// -> the 16 N-rows read by a half-wave hit distinct bank groups.
#define LDS_STRIDE 136

// ---------------------------------------------------------------------------
// Transpose + convert weights: W[k][cin][cout] (f32) -> Wt[k][cout][cin] (bf16)
// ---------------------------------------------------------------------------
__global__ void prep_weights(const float* __restrict__ W,
                             __bf16* __restrict__ Wt, int total) {
  int i = blockIdx.x * 256 + threadIdx.x;
  if (i >= total) return;
  int k  = i / (C_IN * C_OUT);
  int r  = i - k * (C_IN * C_OUT);
  int co = r >> 7;      // cout
  int ci = r & 127;     // cin (fastest in Wt)
  Wt[i] = (__bf16)W[k * (C_IN * C_OUT) + ci * C_OUT + co];
}

// ---------------------------------------------------------------------------
// Convert features to bf16 (4 elements / thread).
// ---------------------------------------------------------------------------
__global__ void feat_to_bf16(const float* __restrict__ f,
                             __bf16* __restrict__ o, long total) {
  long i = ((long)blockIdx.x * 256 + threadIdx.x) * 4;
  if (i >= total) return;
  float4 v = *(const float4*)(f + i);
  union { __bf16 h[4]; uint2 u; } r;
  r.h[0] = (__bf16)v.x; r.h[1] = (__bf16)v.y;
  r.h[2] = (__bf16)v.z; r.h[3] = (__bf16)v.w;
  *(uint2*)(o + i) = r.u;
}

// A 16x32 bf16 fragment (ISA layout): lanes 0-15 hold K 0-7 / 16-23,
// lanes 16-31 hold K 8-15 / 24-31.
__device__ __forceinline__ bf16x16 load_a_frag(const float* __restrict__ row,
                                               int cc, int half) {
  const float* p = row + cc * 32 + half * 8;
  float4 a0 = *(const float4*)(p);
  float4 a1 = *(const float4*)(p + 4);
  float4 a2 = *(const float4*)(p + 16);
  float4 a3 = *(const float4*)(p + 20);
  bf16x16 A;
  A[0]=(__bf16)a0.x;  A[1]=(__bf16)a0.y;  A[2]=(__bf16)a0.z;  A[3]=(__bf16)a0.w;
  A[4]=(__bf16)a1.x;  A[5]=(__bf16)a1.y;  A[6]=(__bf16)a1.z;  A[7]=(__bf16)a1.w;
  A[8]=(__bf16)a2.x;  A[9]=(__bf16)a2.y;  A[10]=(__bf16)a2.z; A[11]=(__bf16)a2.w;
  A[12]=(__bf16)a3.x; A[13]=(__bf16)a3.y; A[14]=(__bf16)a3.z; A[15]=(__bf16)a3.w;
  return A;
}

__device__ __forceinline__ bf16x16 load_a_frag(const __bf16* __restrict__ row,
                                               int cc, int half) {
  const __bf16* p = row + cc * 32 + half * 8;
  union { uint4 u[2]; bf16x16 v; } A;
  A.u[0] = *(const uint4*)(p);        // K half*8 .. +7
  A.u[1] = *(const uint4*)(p + 16);   // K 16+half*8 .. +7
  return A.v;
}

// ---------------------------------------------------------------------------
// Conv GEMM: one wave computes a 32-row x 128-col output super-tile.
// B tile (Wt[k], 32 KB) staged via async global->LDS DMA, double-buffered
// so the DMA of k+1 overlaps the WMMA work of k (ASYNCcnt retires in order).
// ---------------------------------------------------------------------------
template <typename FT>
__global__ __launch_bounds__(256) void conv_wmma(
    const FT*     __restrict__ feat,
    const __bf16* __restrict__ Wt,
    const float*  __restrict__ bias,
    const int*    __restrict__ nb,
    const FT*     __restrict__ zrow,
    float*        __restrict__ out,
    int n)
{
  __shared__ __bf16 Bs[2][C_IN * LDS_STRIDE];   // 2 x 34,816 B

  const int lane  = threadIdx.x & 31;
  const int wave  = threadIdx.x >> 5;
  const int pair  = blockIdx.x * 8 + wave;       // 32-row super-tile id
  const int npair = (n + 31) >> 5;
  const int m     = lane & 15;   // A: M row / B,D: N col
  const int half  = lane >> 4;   // K-half / D row-half selector
  const int row0  = pair * 32 + m;
  const int row1  = row0 + 16;
  const bool act  = (pair < npair);

  floatx8 acc0[8] = {};          // rows [pair*32,      +16)
  floatx8 acc1[8] = {};          // rows [pair*32 + 16, +16)

  // B-staging: thread copies half of one cout row (128 B) via 8 async b128.
  const int sr = threadIdx.x >> 1;   // cout row 0..127
  const int sh = threadIdx.x & 1;    // half of the row
  // LDS byte offsets (generic-address truncation: LDS offset = addr[31:0]).
  const unsigned lb0 = (unsigned)(unsigned long long)(&Bs[0][0])
                       + sr * (LDS_STRIDE * 2) + sh * 128;
  const unsigned lb1 = (unsigned)(unsigned long long)(&Bs[1][0])
                       + sr * (LDS_STRIDE * 2) + sh * 128;

  auto stage = [&](int k, unsigned lbase) {
    const char* g = (const char*)(Wt + (size_t)k * (C_IN * C_OUT))
                    + sr * 256 + sh * 128;
#pragma unroll
    for (int j = 0; j < 8; ++j) {
      unsigned la = lbase + j * 16;
      unsigned long long ga = (unsigned long long)(g + j * 16);
      asm volatile("global_load_async_to_lds_b128 %0, %1, off"
                   :: "v"(la), "v"(ga) : "memory");
    }
  };

  stage(0, lb0);                       // prologue: k=0 into buffer 0

  for (int k = 0; k < KOFF; ++k) {
    __syncthreads();   // all waves done reading buf[(k+1)&1] (iter k-1)
    if (k + 1 < KOFF) {
      stage(k + 1, ((k + 1) & 1) ? lb1 : lb0);
      // Wait for the *older* 8 async ops (stage k); keep stage k+1 in flight.
      asm volatile("s_wait_asynccnt 0x8" ::: "memory");
    } else {
      asm volatile("s_wait_asynccnt 0x0" ::: "memory");
    }
    __syncthreads();   // stage(k) visible to all waves
    const __bf16* Bbuf = &Bs[k & 1][0];

    if (act) {
      int idx0 = (row0 < n) ? nb[row0 * KOFF + k] : -1;
      int idx1 = (row1 < n) ? nb[row1 * KOFF + k] : -1;
      const FT* ar0 = (idx0 >= 0) ? (feat + (size_t)idx0 * C_IN) : zrow;
      const FT* ar1 = (idx1 >= 0) ? (feat + (size_t)idx1 * C_IN) : zrow;

#pragma unroll
      for (int cc = 0; cc < 4; ++cc) {       // cin chunks of 32
        bf16x16 A0 = load_a_frag(ar0, cc, half);
        bf16x16 A1 = load_a_frag(ar1, cc, half);
#pragma unroll
        for (int t = 0; t < 8; ++t) {        // 8 cout tiles of 16
          const __bf16* bp = Bbuf + (t * 16 + m) * LDS_STRIDE + cc * 32 + half * 8;
          union { uint4 u[2]; bf16x16 v; } B;
          B.u[0] = *(const uint4*)(bp);
          B.u[1] = *(const uint4*)(bp + 16);
          acc0[t] = __builtin_amdgcn_wmma_f32_16x16x32_bf16(
              false, A0, false, B.v, (short)0, acc0[t], false, false);
          acc1[t] = __builtin_amdgcn_wmma_f32_16x16x32_bf16(
              false, A1, false, B.v, (short)0, acc1[t], false, false);
        }
      }
    }
  }

  // ---- bias + store (D layout: VGPR r -> M = r + 8*half, lane m -> N) ----
  if (act) {
#pragma unroll
    for (int t = 0; t < 8; ++t) {
      float bc = bias[t * 16 + m];
#pragma unroll
      for (int r = 0; r < 8; ++r) {
        int orow = pair * 32 + half * 8 + r;
        if (orow < n)
          out[(size_t)orow * C_OUT + t * 16 + m] = acc0[t][r] + bc;
        int orow1 = orow + 16;
        if (orow1 < n)
          out[(size_t)orow1 * C_OUT + t * 16 + m] = acc1[t][r] + bc;
      }
    }
  }
}

// ---------------------------------------------------------------------------
// Per-channel sum / sum-of-squares over all active sites.
// ---------------------------------------------------------------------------
__global__ void stats_kernel(const float* __restrict__ out,
                             float* __restrict__ sums, int n) {
  __shared__ float ls[512];
  int c   = threadIdx.x & (C_OUT - 1);
  int sub = threadIdx.x >> 7;                       // 0 or 1
  float s = 0.f, s2 = 0.f;
  for (long r = blockIdx.x * 2 + sub; r < n; r += (long)gridDim.x * 2) {
    float v = out[r * C_OUT + c];
    s += v; s2 += v * v;
  }
  ls[threadIdx.x]       = s;
  ls[256 + threadIdx.x] = s2;
  __syncthreads();
  if (threadIdx.x < C_OUT) {
    atomicAdd(&sums[c],         ls[threadIdx.x]       + ls[threadIdx.x + 128]);
    atomicAdd(&sums[C_OUT + c], ls[256 + threadIdx.x] + ls[256 + threadIdx.x + 128]);
  }
}

// ---------------------------------------------------------------------------
// BatchNorm (batch stats) + LeakyReLU, in place on d_out.
// ---------------------------------------------------------------------------
__global__ void bn_lrelu(float* __restrict__ out,
                         const float* __restrict__ sums,
                         const float* __restrict__ gamma,
                         const float* __restrict__ beta,
                         int n, long total) {
  long i = (long)blockIdx.x * 256 + threadIdx.x;
  if (i >= total) return;
  int c = (int)(i & (C_OUT - 1));
  float invn = 1.0f / (float)n;
  float mean = sums[c] * invn;
  float var  = sums[C_OUT + c] * invn - mean * mean;
  float sc   = rsqrtf(var + EPS_BN) * gamma[c];
  float sh   = beta[c] - mean * sc;
  float y    = out[i] * sc + sh;
  out[i] = (y > 0.0f) ? y : LEAK * y;
}

// ---------------------------------------------------------------------------
extern "C" void kernel_launch(void* const* d_in, const int* in_sizes, int n_in,
                              void* d_out, int out_size, void* d_ws, size_t ws_size,
                              hipStream_t stream) {
  const float* feat  = (const float*)d_in[0];  // [N,128]
  const float* W     = (const float*)d_in[1];  // [27,128,128]
  const float* bias  = (const float*)d_in[2];  // [128]
  const float* gamma = (const float*)d_in[3];  // [128]
  const float* beta  = (const float*)d_in[4];  // [128]
  const int*   nb    = (const int*)d_in[5];    // [N,27]
  float* out = (float*)d_out;

  int n = in_sizes[0] / C_IN;

  char*   ws    = (char*)d_ws;
  float*  sums  = (float*)(ws + WS_SUM);
  __bf16* Wt    = (__bf16*)(ws + WS_WT);
  __bf16* featb = (__bf16*)(ws + WS_FEATB);

  // zero the sums + zero-row every launch (graph-replay safe)
  hipMemsetAsync(ws, 0, 1536, stream);

  int wtotal = KOFF * C_IN * C_OUT;
  prep_weights<<<(wtotal + 255) / 256, 256, 0, stream>>>(W, Wt, wtotal);

  long ftotal = (long)n * C_IN;
  int npair = (n + 31) / 32;
  int cblk  = (npair + 7) / 8;

  // Prefer bf16 pre-converted features (halves gathered A traffic, removes
  // per-k converts) when the workspace is large enough; else convert on the fly.
  if (ws_size >= (size_t)WS_FEATB + (size_t)ftotal * 2) {
    feat_to_bf16<<<(int)((ftotal / 4 + 255) / 256), 256, 0, stream>>>(feat, featb, ftotal);
    conv_wmma<__bf16><<<cblk, 256, 0, stream>>>(
        featb, Wt, bias, nb, (const __bf16*)(ws + WS_ZROW), out, n);
  } else {
    conv_wmma<float><<<cblk, 256, 0, stream>>>(
        feat, Wt, bias, nb, (const float*)(ws + WS_ZROW), out, n);
  }

  stats_kernel<<<512, 256, 0, stream>>>(out, sums, n);

  long total = (long)n * C_OUT;
  bn_lrelu<<<(int)((total + 255) / 256), 256, 0, stream>>>(out, sums, gamma, beta, n, total);
}
